// GatedGN_5342939316457
// MI455X (gfx1250) — matrix operand order repaired and checked by reading, not compiled, not run on hardware
//
#include <hip/hip_runtime.h>
#include <hip/hip_bf16.h>

// ======================= WMMA plumbing (gfx1250) ============================
typedef __attribute__((ext_vector_type(16))) __bf16 v16bf;
typedef __attribute__((ext_vector_type(8)))  float  v8f;

struct __align__(16) U4 { unsigned int x, y, z, w; };
union Frag { v16bf v; U4 q[2]; };

__device__ __forceinline__ unsigned short f2bf(float f) {
  union { float f; unsigned u; } v; v.f = f;
  unsigned r = v.u + 0x7FFFu + ((v.u >> 16) & 1u);   // round-to-nearest-even
  return (unsigned short)(r >> 16);
}

__device__ __forceinline__ v8f vzero8() {
  v8f z;
#pragma unroll
  for (int i = 0; i < 8; ++i) z[i] = 0.f;
  return z;
}

__device__ __forceinline__ float sigm(float x) { return 1.f / (1.f + __expf(-x)); }

// 64x128 tile GEMM core: A (64x128 bf16, row-major LDS), W (128x128 bf16,
// TRANSPOSED [out][in] in LDS). Wave covers row-tile rt (16 rows) and column
// half ch (4x 16-col tiles). Emits v_wmma_f32_16x16x32_bf16.
__device__ __forceinline__ void mm64x128(const unsigned short* __restrict__ Abuf,
                                         const unsigned short* __restrict__ Wbuf,
                                         int rt, int ch, int lane, v8f acc[4]) {
  const int m = lane & 15, half = lane >> 4;
#pragma unroll
  for (int kc = 0; kc < 4; ++kc) {
    Frag a;
    const unsigned short* ap = Abuf + (rt * 16 + m) * 128 + kc * 32 + 8 * half;
    a.q[0] = *(const U4*)ap;          // k = kc*32 + 8h .. +7
    a.q[1] = *(const U4*)(ap + 16);   // k = kc*32 + 16 + 8h .. +7
#pragma unroll
    for (int t = 0; t < 4; ++t) {
      const int col = (ch * 4 + t) * 16 + m;
      Frag b;
      const unsigned short* bp = Wbuf + col * 128 + kc * 32 + 8 * half;
      b.q[0] = *(const U4*)bp;        // k = kc*32 + 8h .. +7
      b.q[1] = *(const U4*)(bp + 16); // k = kc*32 + 16 + 8h .. +7
      acc[t] = __builtin_amdgcn_wmma_f32_16x16x32_bf16(
          false, a.v, false, b.v, (short)0, acc[t], false, false);
    }
  }
}

__device__ __forceinline__ void stage_w(const unsigned short* __restrict__ src,
                                        unsigned short* __restrict__ Ws, int tid) {
  const U4* s = (const U4*)src;
  U4* d = (U4*)Ws;
  for (int i = tid; i < 2048; i += 256) d[i] = s[i];   // 32KB
}

// ======================= generic GEMM: Y = act(X@W + b [+Y]) =================
// X: fp32, (R/xRowDiv) x 128 ; output row r reads X row r/xRowDiv (broadcast).
// Wt: bf16 transposed 128x128. grid.x = R/64, block = 256.
__global__ __launch_bounds__(256) void gemm_wmma_h128(
    const float* __restrict__ X, int xRowDiv,
    const unsigned short* __restrict__ Wt,
    const float* __restrict__ bias,
    float* __restrict__ Y, int relu, int accum) {
  __shared__ unsigned short Xs[64 * 128];
  __shared__ unsigned short Ws[128 * 128];
  const int tid = threadIdx.x;
  const int r0 = blockIdx.x * 64;
  stage_w(Wt, Ws, tid);
  for (int i = tid; i < 64 * 128; i += 256) {
    const int r = i >> 7, c = i & 127;
    Xs[i] = f2bf(X[(size_t)((r0 + r) / xRowDiv) * 128 + c]);
  }
  __syncthreads();
  const int wave = tid >> 5, lane = tid & 31;
  const int rt = wave >> 1, ch = wave & 1;
  v8f acc[4];
#pragma unroll
  for (int t = 0; t < 4; ++t) acc[t] = vzero8();
  mm64x128(Xs, Ws, rt, ch, lane, acc);
  const int nIdx = lane & 15, half = lane >> 4;
  const int rbase = r0 + rt * 16 + 8 * half;
#pragma unroll
  for (int t = 0; t < 4; ++t) {
    const int col = (ch * 4 + t) * 16 + nIdx;
    const float bv = bias ? bias[col] : 0.f;
#pragma unroll
    for (int v = 0; v < 8; ++v) {
      const size_t o = (size_t)(rbase + v) * 128 + col;
      float val = acc[t][v] + bv;
      if (accum) val += Y[o];
      if (relu) val = fmaxf(val, 0.f);
      Y[o] = val;
    }
  }
}

// ======================= fused edge kernel ==================================
// One block per graph n. Handles all 64 edges (i,j), H=128:
//   t   = relu(Q1[i]+Q2[j]+b_mn0)             (precomputed factored pair)
//   t2  = relu(t @ W1 + b_mn1)                (WMMA)
//   t3  = t2 @ W2                             (WMMA)
//   u   = sigmoid(Amu + e@Wue + b_ue + P1[i] + P2[j] + b_mun)   (WMMA)
//   e'  = (1-u)*e + u*tanh(Amg + 2*(t3 + b_mn2))
//   edgesOut[n,i] = sum_{j!=i} e' ; esOut[n] = sum_{i,j} e'   (deterministic)
__global__ __launch_bounds__(256) void edge_fused(
    float* __restrict__ E,
    const float* __restrict__ Q1, const float* __restrict__ Q2,
    const float* __restrict__ P1, const float* __restrict__ P2,
    const float* __restrict__ Amu, const float* __restrict__ Amg,
    const unsigned short* __restrict__ W1t, const unsigned short* __restrict__ W2t,
    const unsigned short* __restrict__ Wuet,
    const float* __restrict__ b_mn0, const float* __restrict__ b_mn1,
    const float* __restrict__ b_mn2, const float* __restrict__ b_mun,
    const float* __restrict__ b_ue,
    float* __restrict__ edgesOut, float* __restrict__ esOut) {
  __shared__ unsigned short ebf[64 * 128];   // 16KB
  __shared__ unsigned short tbf[64 * 128];   // 16KB
  __shared__ unsigned short Ws[128 * 128];   // 32KB (aliased as esum later)
  float* esum = (float*)Ws;                  // 8x128 f32, reused after Wue
  const int n = blockIdx.x, tid = threadIdx.x;
  const float* Erow = E + (size_t)n * 8192;
  for (int i = tid; i < 8192; i += 256) {
    const int r = i >> 7, c = i & 127;
    ebf[i] = f2bf(Erow[i]);
    const int ii = r >> 3, jj = r & 7;
    const float tv = Q1[((size_t)n * 8 + ii) * 128 + c] +
                     Q2[((size_t)n * 8 + jj) * 128 + c] + b_mn0[c];
    tbf[i] = f2bf(fmaxf(tv, 0.f));
  }
  stage_w(W1t, Ws, tid);
  __syncthreads();

  const int wave = tid >> 5, lane = tid & 31;
  const int rt = wave >> 1, ch = wave & 1;
  const int m = lane & 15, half = lane >> 4;
  const int iNode = 2 * rt + half;          // each lane owns one node i
  const int rloc = rt * 16 + 8 * half;      // == iNode*8

  // ---- t2 = relu(t @ W1 + b1) -------------------------------------------
  v8f a2[4];
#pragma unroll
  for (int t = 0; t < 4; ++t) a2[t] = vzero8();
  mm64x128(tbf, Ws, rt, ch, lane, a2);
  __syncthreads();
#pragma unroll
  for (int t = 0; t < 4; ++t) {
    const int col = (ch * 4 + t) * 16 + m;
    const float bv = b_mn1[col];
#pragma unroll
    for (int v = 0; v < 8; ++v)
      tbf[(rloc + v) * 128 + col] = f2bf(fmaxf(a2[t][v] + bv, 0.f));
  }
  stage_w(W2t, Ws, tid);
  __syncthreads();

  // ---- t3 = t2 @ W2 ------------------------------------------------------
  v8f a3[4];
#pragma unroll
  for (int t = 0; t < 4; ++t) a3[t] = vzero8();
  mm64x128(tbf, Ws, rt, ch, lane, a3);
  __syncthreads();
  stage_w(Wuet, Ws, tid);
  __syncthreads();

  // ---- u_lin = e @ Wue ---------------------------------------------------
  v8f au[4];
#pragma unroll
  for (int t = 0; t < 4; ++t) au[t] = vzero8();
  mm64x128(ebf, Ws, rt, ch, lane, au);

  // ---- gating epilogue ---------------------------------------------------
  float fullsum[4];
#pragma unroll
  for (int t = 0; t < 4; ++t) {
    const int col = (ch * 4 + t) * 16 + m;
    const float base_u = Amu[(size_t)n * 128 + col] + b_ue[col] + b_mun[col] +
                         P1[((size_t)n * 8 + iNode) * 128 + col];
    const float amg = Amg[(size_t)n * 128 + col];
    const float b2v = b_mn2[col];
    float ssum = 0.f, diag = 0.f;
#pragma unroll
    for (int v = 0; v < 8; ++v) {        // v == j
      const float p2 = P2[((size_t)n * 8 + v) * 128 + col];
      const float u = sigm(base_u + au[t][v] + p2);
      const float ne = tanhf(amg + 2.f * (a3[t][v] + b2v));
      const size_t gi = (size_t)n * 8192 + (size_t)(rloc + v) * 128 + col;
      const float env = (1.f - u) * E[gi] + u * ne;
      E[gi] = env;
      ssum += env;
      if (v == iNode) diag = env;
    }
    edgesOut[((size_t)n * 8 + iNode) * 128 + col] = ssum - diag;
    fullsum[t] = ssum;
  }
  __syncthreads();                         // Ws (Wue) reads complete
#pragma unroll
  for (int t = 0; t < 4; ++t)
    esum[iNode * 128 + (ch * 4 + t) * 16 + m] = fullsum[t];
  __syncthreads();
  if (tid < 128) {
    float s = 0.f;
#pragma unroll
    for (int i = 0; i < 8; ++i) s += esum[i * 128 + tid];
    esOut[(size_t)n * 128 + tid] = s;
  }
}

// ======================= small helper kernels ===============================
__global__ void lin14_k(const float* __restrict__ X, const float* __restrict__ W,
                        const float* __restrict__ b, float* __restrict__ Y,
                        int relu, int accum) {
  __shared__ float xs[14];
  const int row = blockIdx.x, c = threadIdx.x;
  if (c < 14) xs[c] = X[(size_t)row * 14 + c];
  __syncthreads();
  float s = b[c];
#pragma unroll
  for (int i = 0; i < 14; ++i) s += xs[i] * W[i * 128 + c];
  const size_t o = (size_t)row * 128 + c;
  if (accum) s += Y[o];
  if (relu) s = fmaxf(s, 0.f);
  Y[o] = s;
}

__global__ void fill_bias_k(float* __restrict__ Y, const float* __restrict__ b, int total) {
  const int i = blockIdx.x * 256 + threadIdx.x;
  if (i < total) Y[i] = b[i & 127];
}
__global__ void sigmoid_k(float* __restrict__ Y, int total) {
  const int i = blockIdx.x * 256 + threadIdx.x;
  if (i < total) Y[i] = sigm(Y[i]);
}
__global__ void mul_k(float* __restrict__ C, const float* __restrict__ A,
                      const float* __restrict__ B, int total) {
  const int i = blockIdx.x * 256 + threadIdx.x;
  if (i < total) C[i] = A[i] * B[i];
}
__global__ void gate_k(float* __restrict__ H, const float* __restrict__ U,
                       const float* __restrict__ S, int total) {
  const int i = blockIdx.x * 256 + threadIdx.x;
  if (i < total) H[i] = (1.f - U[i]) * H[i] + U[i] * tanhf(S[i]);
}
__global__ void sumk_k(float* __restrict__ HS, const float* __restrict__ Hh) {
  const int n = blockIdx.x, c = threadIdx.x;
  float s = 0.f;
#pragma unroll
  for (int i = 0; i < 8; ++i) s += Hh[((size_t)n * 8 + i) * 128 + c];
  HS[(size_t)n * 128 + c] = s;
}
__global__ void out_head_k(const float* __restrict__ Z, const float* __restrict__ w,
                           const float* __restrict__ b, float* __restrict__ out) {
  __shared__ float red[128];
  const int n = blockIdx.x, t = threadIdx.x;
  red[t] = Z[(size_t)n * 128 + t] * w[t];
  __syncthreads();
  for (int s = 64; s > 0; s >>= 1) {
    if (t < s) red[t] += red[t + s];
    __syncthreads();
  }
  if (t == 0) out[n] = sigm(red[0] + b[0]);
}

// fp32 -> transposed bf16 weight chunks (128x128 each)
struct WSrcTab { const float* p[45]; };
__global__ __launch_bounds__(256) void conv_weights_k(WSrcTab tab, unsigned short* dst) {
  const float* s = tab.p[blockIdx.x];
  unsigned short* d = dst + (size_t)blockIdx.x * 16384;
  for (int idx = threadIdx.x; idx < 16384; idx += 256) {
    const int i = idx >> 7, o = idx & 127;
    d[o * 128 + i] = f2bf(s[idx]);
  }
}

// ======================= host orchestration =================================
extern "C" void kernel_launch(void* const* d_in, const int* in_sizes, int n_in,
                              void* d_out, int out_size, void* d_ws, size_t ws_size,
                              hipStream_t stream) {
  (void)in_sizes; (void)n_in; (void)out_size; (void)ws_size;
  const int N = 2048, H = 128, RN = 2048 * 8, KITERS = 2;  // k==2 (launch-shape)

  const float* towers = (const float*)d_in[0];
  int pi = 2;
  struct Lin { const float* w; const float* b; };
  struct Mlp { Lin l0, l1, l2; };
  auto LIN = [&]() { Lin l{(const float*)d_in[pi], (const float*)d_in[pi + 1]}; pi += 2; return l; };
  auto MLPp = [&]() { Mlp m; m.l0 = LIN(); m.l1 = LIN(); m.l2 = LIN(); return m; };
  Lin E_n = LIN(), E_e = LIN(), E_g = LIN();
  Mlp Mn = MLPp(), Me = MLPp(), Mg = MLPp();
  Lin Mun = LIN(), Mue = LIN(), Mug = LIN();
  Lin Mrn = LIN(), Mre = LIN(), Mrg = LIN();
  Mlp Uf = MLPp(), Un = MLPp(), Ue = MLPp(), Ug = MLPp();
  Lin Uuf = LIN(), Uun = LIN(), Uue = LIN(), Uug = LIN();
  Lin Urf = LIN(), Urn = LIN(), Ure = LIN(), Urg = LIN();
  Mlp Gn = MLPp(), Ge = MLPp(), Gg = MLPp();
  Lin Gupn = LIN(), Gupe = LIN(), Gupg = LIN();
  Lin Grn = LIN(), Gre = LIN(), Grg = LIN();
  Mlp O = MLPp();
  (void)Me; (void)Mrn; (void)Mre; (void)Mrg;  // dead branches in the reference

  // ---- workspace carve ----
  size_t off = 0;
  auto carve = [&](size_t bytes) { void* p = (char*)d_ws + off; off += (bytes + 255) & ~(size_t)255; return p; };
  unsigned short* WB = (unsigned short*)carve((size_t)45 * 16384 * 2);
  float* hbuf = (float*)carve((size_t)RN * H * 4);
  float* ebuf = (float*)carve((size_t)N * 64 * H * 4);
  float* B[7];  for (int i = 0; i < 7; ++i) B[i] = (float*)carve((size_t)RN * H * 4);
  float* S[8];  for (int i = 0; i < 8; ++i) S[i] = (float*)carve((size_t)N * H * 4);
  float* gbuf = (float*)carve((size_t)N * H * 4);

  // ---- register bf16 weight chunks (order fixed) ----
  WSrcTab tab; int slot = 0;
  auto reg = [&](const float* w) { tab.p[slot] = w; return slot++; };
  const int sMn0a = reg(Mn.l0.w), sMn0b = reg(Mn.l0.w + 16384);
  const int sMn1 = reg(Mn.l1.w), sMn2 = reg(Mn.l2.w);
  const int sMg0 = reg(Mg.l0.w), sMg1 = reg(Mg.l1.w), sMg2 = reg(Mg.l2.w);
  const int sMuna = reg(Mun.w), sMunb = reg(Mun.w + 16384);
  const int sMue = reg(Mue.w), sMug = reg(Mug.w);
  const int sUn0 = reg(Un.l0.w), sUn1 = reg(Un.l1.w), sUn2 = reg(Un.l2.w);
  const int sUe0 = reg(Ue.l0.w), sUe1 = reg(Ue.l1.w), sUe2 = reg(Ue.l2.w);
  const int sUg0 = reg(Ug.l0.w), sUg1 = reg(Ug.l1.w), sUg2 = reg(Ug.l2.w);
  const int sUf1 = reg(Uf.l1.w), sUf2 = reg(Uf.l2.w);
  const int sUun = reg(Uun.w), sUue = reg(Uue.w), sUug = reg(Uug.w);
  const int sUrn = reg(Urn.w), sUre = reg(Ure.w), sUrg = reg(Urg.w);
  const int sGn0 = reg(Gn.l0.w), sGn1 = reg(Gn.l1.w), sGn2 = reg(Gn.l2.w);
  const int sGe0 = reg(Ge.l0.w), sGe1 = reg(Ge.l1.w), sGe2 = reg(Ge.l2.w);
  const int sGg0 = reg(Gg.l0.w), sGg1 = reg(Gg.l1.w), sGg2 = reg(Gg.l2.w);
  const int sGupn = reg(Gupn.w), sGupe = reg(Gupe.w), sGupg = reg(Gupg.w);
  const int sGrn = reg(Grn.w), sGre = reg(Gre.w), sGrg = reg(Grg.w);
  const int sO0 = reg(O.l0.w), sO1 = reg(O.l1.w);
  conv_weights_k<<<dim3(slot), dim3(256), 0, stream>>>(tab, WB);

  auto Wp = [&](int s) { return WB + (size_t)s * 16384; };
  auto gemm = [&](float* Y, const float* X, int rowdiv, int ws, const float* bias,
                  int R, int relu, int accum) {
    gemm_wmma_h128<<<dim3(R / 64), dim3(256), 0, stream>>>(X, rowdiv, Wp(ws), bias, Y, relu, accum);
  };
  auto ew = [&](int total) { return dim3((total + 255) / 256); };

  // ---- init: h = towers@E_n + b ; e = bcast(E_e.b) ; g = bcast(E_g.b) ----
  lin14_k<<<dim3(RN), dim3(128), 0, stream>>>(towers, E_n.w, E_n.b, hbuf, 0, 0);
  fill_bias_k<<<ew(N * 64 * H), dim3(256), 0, stream>>>(ebuf, E_e.b, N * 64 * H);
  fill_bias_k<<<ew(N * H), dim3(256), 0, stream>>>(gbuf, E_g.b, N * H);

  for (int it = 0; it < KITERS; ++it) {
    // ---- global-level projections ----
    gemm(S[0], gbuf, 1, sMug, Mug.b, N, 0, 0);                 // A_mu
    gemm(S[1], gbuf, 1, sMg0, Mg.l0.b, N, 1, 0);
    gemm(S[2], S[1], 1, sMg1, Mg.l1.b, N, 1, 0);
    gemm(S[3], S[2], 1, sMg2, Mg.l2.b, N, 0, 0);               // A_mg
    // ---- node-level factored pair projections ----
    gemm(B[0], hbuf, 1, sMuna, nullptr, RN, 0, 0);             // P1
    gemm(B[1], hbuf, 1, sMunb, nullptr, RN, 0, 0);             // P2
    gemm(B[2], hbuf, 1, sMn0a, nullptr, RN, 0, 0);             // Q1
    gemm(B[3], hbuf, 1, sMn0b, nullptr, RN, 0, 0);             // Q2
    // ---- fused edge update + edge sums ----
    edge_fused<<<dim3(N), dim3(256), 0, stream>>>(
        ebuf, B[2], B[3], B[0], B[1], S[0], S[3],
        Wp(sMn1), Wp(sMn2), Wp(sMue),
        Mn.l0.b, Mn.l1.b, Mn.l2.b, Mun.b, Mue.b, B[4], S[4]);  // edges=B4, es=S4
    // ---- node fn ----
    gemm(B[0], gbuf, 8, sUrg, Urg.b, RN, 0, 0);                // reset
    gemm(B[0], B[4], 1, sUre, Ure.b, RN, 0, 1);
    gemm(B[0], hbuf, 1, sUrn, Urn.b, RN, 0, 1);
    lin14_k<<<dim3(RN), dim3(128), 0, stream>>>(towers, Urf.w, Urf.b, B[0], 0, 1);
    sigmoid_k<<<ew(RN * H), dim3(256), 0, stream>>>(B[0], RN * H);
    gemm(B[1], gbuf, 8, sUug, Uug.b, RN, 0, 0);                // upd_n
    gemm(B[1], B[4], 1, sUue, Uue.b, RN, 0, 1);
    gemm(B[1], hbuf, 1, sUun, Uun.b, RN, 0, 1);
    lin14_k<<<dim3(RN), dim3(128), 0, stream>>>(towers, Uuf.w, Uuf.b, B[1], 0, 1);
    sigmoid_k<<<ew(RN * H), dim3(256), 0, stream>>>(B[1], RN * H);
    mul_k<<<ew(RN * H), dim3(256), 0, stream>>>(B[2], B[0], hbuf, RN * H);  // reset*h
    lin14_k<<<dim3(RN), dim3(128), 0, stream>>>(towers, Uf.l0.w, Uf.l0.b, B[5], 1, 0);
    gemm(B[6], B[5], 1, sUf1, Uf.l1.b, RN, 1, 0);
    gemm(B[3], B[6], 1, sUf2, Uf.l2.b, RN, 0, 0);              // S = mlp(U_feats)
    gemm(B[5], B[2], 1, sUn0, Un.l0.b, RN, 1, 0);
    gemm(B[6], B[5], 1, sUn1, Un.l1.b, RN, 1, 0);
    gemm(B[3], B[6], 1, sUn2, Un.l2.b, RN, 0, 1);              // += mlp(U_node)
    gemm(B[5], B[4], 1, sUe0, Ue.l0.b, RN, 1, 0);
    gemm(B[6], B[5], 1, sUe1, Ue.l1.b, RN, 1, 0);
    gemm(B[3], B[6], 1, sUe2, Ue.l2.b, RN, 0, 1);              // += mlp(U_edges)
    gemm(S[5], gbuf, 1, sUg0, Ug.l0.b, N, 1, 0);
    gemm(S[6], S[5], 1, sUg1, Ug.l1.b, N, 1, 0);
    gemm(B[3], S[6], 8, sUg2, Ug.l2.b, RN, 0, 1);              // += bcast mlp(U_global)
    gate_k<<<ew(RN * H), dim3(256), 0, stream>>>(hbuf, B[1], B[3], RN * H);
    // ---- global fn ----
    sumk_k<<<dim3(N), dim3(128), 0, stream>>>(S[5], hbuf);     // hs
    gemm(S[6], gbuf, 1, sGrg, Grg.b, N, 0, 0);                 // r_g
    gemm(S[6], S[5], 1, sGrn, Grn.b, N, 0, 1);
    gemm(S[6], S[4], 1, sGre, Gre.b, N, 0, 1);
    sigmoid_k<<<ew(N * H), dim3(256), 0, stream>>>(S[6], N * H);
    gemm(S[7], gbuf, 1, sGupg, Gupg.b, N, 0, 0);               // u_g (all on g, faithful)
    gemm(S[7], gbuf, 1, sGupn, Gupn.b, N, 0, 1);
    gemm(S[7], gbuf, 1, sGupe, Gupe.b, N, 0, 1);
    sigmoid_k<<<ew(N * H), dim3(256), 0, stream>>>(S[7], N * H);
    mul_k<<<ew(N * H), dim3(256), 0, stream>>>(S[0], S[6], gbuf, N * H);    // r_g*g
    gemm(S[1], S[0], 1, sGg0, Gg.l0.b, N, 1, 0);
    gemm(S[2], S[1], 1, sGg1, Gg.l1.b, N, 1, 0);
    gemm(S[3], S[2], 1, sGg2, Gg.l2.b, N, 0, 0);               // gS = mlp(G_global)
    gemm(S[1], S[4], 1, sGe0, Ge.l0.b, N, 1, 0);
    gemm(S[2], S[1], 1, sGe1, Ge.l1.b, N, 1, 0);
    gemm(S[3], S[2], 1, sGe2, Ge.l2.b, N, 0, 1);               // += mlp(G_edges)
    gemm(S[1], S[5], 1, sGn0, Gn.l0.b, N, 1, 0);
    gemm(S[2], S[1], 1, sGn1, Gn.l1.b, N, 1, 0);
    gemm(S[3], S[2], 1, sGn2, Gn.l2.b, N, 0, 1);               // += mlp(G_nodes)
    gate_k<<<ew(N * H), dim3(256), 0, stream>>>(gbuf, S[7], S[3], N * H);
  }
  // ---- output head ----
  gemm(S[1], gbuf, 1, sO0, O.l0.b, N, 1, 0);
  gemm(S[2], S[1], 1, sO1, O.l1.b, N, 1, 0);
  out_head_k<<<dim3(N), dim3(128), 0, stream>>>(S[2], O.l2.w, O.l2.b, (float*)d_out);
}
